// MultiHeadAttention_63969242906711
// MI455X (gfx1250) — matrix operand order repaired and checked by reading, not compiled, not run on hardware
//
#include <hip/hip_runtime.h>
#include <hip/hip_bf16.h>
#include <math.h>

// ---------------------------------------------------------------------------
// CDNA5 (gfx1250) bf16 WMMA multi-head attention
//   B=8, L=1024, D=1024, H=16, HD=64; softmax over QUERY axis.
// ---------------------------------------------------------------------------

typedef __attribute__((ext_vector_type(16))) __bf16 v16bf;
typedef __attribute__((ext_vector_type(8)))  float  v8f;

union BF16Frag {
    v16bf        v;
    unsigned int d[8];
};

static __device__ inline unsigned short f2bf(float f) {
    unsigned int u = __float_as_uint(f);
    u += 0x7FFFu + ((u >> 16) & 1u);   // round-to-nearest-even
    return (unsigned short)(u >> 16);
}

static __device__ inline unsigned int f2bf_pk(float a, float b) {
    return (unsigned int)f2bf(a) | ((unsigned int)f2bf(b) << 16);
}

static __device__ inline v8f v8f_zero() {
    v8f z;
#pragma unroll
    for (int i = 0; i < 8; ++i) z[i] = 0.0f;
    return z;
}

static __device__ inline v8f wmma_bf16(const BF16Frag& a, const BF16Frag& b, v8f c) {
    return __builtin_amdgcn_wmma_f32_16x16x32_bf16(
        /*neg_a=*/false, a.v, /*neg_b=*/false, b.v,
        /*c_mod=*/(short)0, c, /*reuse_a=*/false, /*reuse_b=*/false);
}

// CDNA5 async global->LDS copy (4 bytes per lane), tracked with ASYNCcnt.
static __device__ inline void async_ld_b32(const void* g, void* l) {
    unsigned int loff = (unsigned int)(unsigned long long)l;  // low 32 bits = LDS offset
    asm volatile("global_load_async_to_lds_b32 %0, %1, off"
                 :: "v"(loff), "v"(g) : "memory");
}
static __device__ inline void wait_async0() {
    asm volatile("s_wait_asynccnt 0x0" ::: "memory");
}

// ---------------------------------------------------------------------------
// GEMM + bias:  C[M,N] = A[M,K] * W[K,N] + bias[N]
//   A_F32:  A is fp32 (converted to bf16 while staging); else bf16 and the
//           A tile is staged with GLOBAL_LOAD_ASYNC_TO_LDS_B32.
//   OUT_F32: C written fp32, else bf16
// Block: 256 threads (8 waves), tile 128x128, K-step 32; each wave computes
// 32x64 (2x4 WMMA tiles) -> 8 v_wmma per wave per k-step.
// ---------------------------------------------------------------------------
template <bool A_F32, bool OUT_F32>
__global__ __launch_bounds__(256) void gemm_bias_kernel(
    const void* __restrict__ A_, const float* __restrict__ W,
    const float* __restrict__ bias, void* __restrict__ out_,
    int M, int N, int K)
{
    __shared__ unsigned short As[128][36];   // [m][k]   stride 36 -> b32 pairs 4B-aligned
    __shared__ unsigned short Bs[128][34];   // [n][k]   (W transposed on staging)

    const int t    = threadIdx.x;
    const int lane = t & 31;
    const int w    = t >> 5;       // 0..7
    const int wr   = w >> 1;       // wave row (0..3), 32 rows each
    const int wc   = w & 1;        // wave col (0..1), 64 cols each
    const int lo   = lane & 15;
    const int hi   = lane >> 4;
    const int blkM = blockIdx.y * 128;
    const int blkN = blockIdx.x * 128;

    const float*          Af = (const float*)A_;
    const unsigned short* Ah = (const unsigned short*)A_;

    v8f acc[2][4];
#pragma unroll
    for (int i = 0; i < 2; ++i)
#pragma unroll
        for (int j = 0; j < 4; ++j) acc[i][j] = v8f_zero();

    for (int kk = 0; kk < K; kk += 32) {
        // ---- stage A tile (128x32) ----
        if (A_F32) {
            // fp32 -> bf16 conversion, float4 loads, packed b32 LDS stores
#pragma unroll
            for (int i = 0; i < 4; ++i) {
                int e  = t + i * 256;           // 1024 float4s
                int r  = e >> 3, c4 = (e & 7) * 4;
                const float4 fv = *(const float4*)&Af[(size_t)(blkM + r) * K + kk + c4];
                *(unsigned int*)&As[r][c4]     = f2bf_pk(fv.x, fv.y);
                *(unsigned int*)&As[r][c4 + 2] = f2bf_pk(fv.z, fv.w);
            }
        } else {
            // bf16 already: raw async DMA into LDS (one dword pair per lane)
#pragma unroll
            for (int i = 0; i < 4; ++i) {
                int e  = t + i * 256;           // 1024 dwords
                int r  = e >> 4, c2 = (e & 15) * 2;
                async_ld_b32(Ah + (size_t)(blkM + r) * K + kk + c2, &As[r][c2]);
            }
        }
        // ---- stage W tile (32x128) transposed -> Bs[n][k] ----
#pragma unroll
        for (int i = 0; i < 4; ++i) {
            int e  = t + i * 256;               // 1024 float4s
            int kq = e >> 5, n4 = (e & 31) * 4;
            const float4 fv = *(const float4*)&W[(size_t)(kk + kq) * N + blkN + n4];
            Bs[n4 + 0][kq] = f2bf(fv.x);
            Bs[n4 + 1][kq] = f2bf(fv.y);
            Bs[n4 + 2][kq] = f2bf(fv.z);
            Bs[n4 + 3][kq] = f2bf(fv.w);
        }
        if (kk + 32 < K)  // hint next W tile toward L2
            __builtin_prefetch(&W[(size_t)(kk + 32) * N + blkN + (t & 127)], 0, 1);
        if (!A_F32) wait_async0();
        __syncthreads();

        // ---- load fragments ----
        BF16Frag a[2], b[4];
#pragma unroll
        for (int s = 0; s < 2; ++s) {
            int row = wr * 32 + s * 16 + lo;
            int kb  = hi * 8;
#pragma unroll
            for (int j = 0; j < 4; ++j) {
                a[s].d[j]     = *(const unsigned int*)&As[row][kb + 2 * j];
                a[s].d[j + 4] = *(const unsigned int*)&As[row][kb + 16 + 2 * j];
            }
        }
#pragma unroll
        for (int s = 0; s < 4; ++s) {
            int col = wc * 64 + s * 16 + lo;
            int kb  = hi * 8;
#pragma unroll
            for (int j = 0; j < 8; ++j)
                b[s].d[j] = *(const unsigned int*)&Bs[col][kb + 2 * j + (j >> 2) * 8];
        }
        // ---- 2x4 WMMA ----
#pragma unroll
        for (int i = 0; i < 2; ++i)
#pragma unroll
            for (int j = 0; j < 4; ++j)
                acc[i][j] = wmma_bf16(a[i], b[j], acc[i][j]);

        __syncthreads();
    }

    // ---- epilogue: bias add, store ----
#pragma unroll
    for (int i = 0; i < 2; ++i) {
#pragma unroll
        for (int j = 0; j < 4; ++j) {
            int   n  = blkN + wc * 64 + j * 16 + lo;
            float bz = bias[n];
#pragma unroll
            for (int r = 0; r < 8; ++r) {
                int   m = blkM + wr * 32 + i * 16 + hi * 8 + r;
                float v = acc[i][j][r] + bz;
                if (OUT_F32) ((float*)out_)[(size_t)m * N + n] = v;
                else ((unsigned short*)out_)[(size_t)m * N + n] = f2bf(v);
            }
        }
    }
}

// ---------------------------------------------------------------------------
// Pass 1: per-column (query-axis) softmax stats.
// Grid (L/64, B*H). Each wave owns 16 k-columns; K fragments stay in registers
// across the whole q loop (64 q-tiles x 2 WMMA = 128 WMMA per wave).
// ---------------------------------------------------------------------------
__global__ __launch_bounds__(128) void attn_stats_kernel(
    const unsigned short* __restrict__ Qbf, const unsigned short* __restrict__ Kbf,
    float* __restrict__ mbuf, float* __restrict__ sbuf)
{
    const int bh = blockIdx.y;
    const int b = bh >> 4, h = bh & 15;
    const int w = threadIdx.x >> 5, lane = threadIdx.x & 31;
    const int lo = lane & 15, hi = lane >> 4;
    const int kc0 = blockIdx.x * 64 + w * 16;
    const float scale = 0.03125f;   // 1/sqrt(D)

    // B fragments of K^T (N=k column, reduction dim = d), both 32-d halves
    BF16Frag bk[2];
    {
        int kcol = kc0 + lo;
        const unsigned short* kp = Kbf + (size_t)(b * 1024 + kcol) * 1024 + h * 64;
        int kb = hi * 8;
#pragma unroll
        for (int half = 0; half < 2; ++half)
#pragma unroll
            for (int j = 0; j < 8; ++j)
                bk[half].d[j] = *(const unsigned int*)(kp + half * 32 + kb + 2 * j + (j >> 2) * 8);
    }

    float m = -3.0e38f, s = 0.0f;
    for (int qt = 0; qt < 64; ++qt) {
        BF16Frag a[2];
        int q = qt * 16 + lo;
        const unsigned short* qp = Qbf + (size_t)(b * 1024 + q) * 1024 + h * 64;
        int kb = hi * 8;
#pragma unroll
        for (int half = 0; half < 2; ++half)
#pragma unroll
            for (int j = 0; j < 4; ++j) {
                a[half].d[j]     = *(const unsigned int*)(qp + half * 32 + kb + 2 * j);
                a[half].d[j + 4] = *(const unsigned int*)(qp + half * 32 + kb + 16 + 2 * j);
            }
        v8f c = v8f_zero();
        c = wmma_bf16(a[0], bk[0], c);
        c = wmma_bf16(a[1], bk[1], c);

        float x[8], tm = -3.0e38f;
#pragma unroll
        for (int j = 0; j < 8; ++j) { x[j] = c[j] * scale; tm = fmaxf(tm, x[j]); }
        float mn = fmaxf(m, tm);
        float ac = 0.0f;
#pragma unroll
        for (int j = 0; j < 8; ++j) ac += __expf(x[j] - mn);
        s = s * __expf(m - mn) + ac;
        m = mn;
    }
    // merge the two 8-row halves of each column (lane <-> lane^16)
    float pm = __shfl_xor(m, 16, 32);
    float ps = __shfl_xor(s, 16, 32);
    float mn = fmaxf(m, pm);
    s = s * __expf(m - mn) + ps * __expf(pm - mn);
    m = mn;
    if (lane < 16) {
        mbuf[(size_t)bh * 1024 + kc0 + lane] = m;
        sbuf[(size_t)bh * 1024 + kc0 + lane] = s;
    }
}

// ---------------------------------------------------------------------------
// Pass 2: O = softmax_q(S) @ V.  Grid (L/64, B*H); wave owns 16 q-rows.
// Recompute S per 32-k chunk, normalize with (m,s), repack P via LDS,
// accumulate O[16x64] with 4 WMMA per chunk.
// ---------------------------------------------------------------------------
__global__ __launch_bounds__(128) void attn_out_kernel(
    const unsigned short* __restrict__ Qbf, const unsigned short* __restrict__ Kbf,
    const unsigned short* __restrict__ Vbf,
    const float* __restrict__ mbuf, const float* __restrict__ sbuf,
    unsigned short* __restrict__ ctx)
{
    __shared__ unsigned short Vs[64][34];      // [d][k] (transposed V chunk)
    __shared__ unsigned short Ps[4][16][34];   // per-wave P tile [q][k]

    const int bh = blockIdx.y;
    const int b = bh >> 4, h = bh & 15;
    const int w = threadIdx.x >> 5, lane = threadIdx.x & 31;
    const int lo = lane & 15, hi = lane >> 4;
    const int q0 = blockIdx.x * 64 + w * 16;
    const float scale = 0.03125f;

    // Q fragments resident for the whole k loop
    BF16Frag aq[2];
    {
        int q = q0 + lo;
        const unsigned short* qp = Qbf + (size_t)(b * 1024 + q) * 1024 + h * 64;
        int kb = hi * 8;
#pragma unroll
        for (int half = 0; half < 2; ++half)
#pragma unroll
            for (int j = 0; j < 4; ++j) {
                aq[half].d[j]     = *(const unsigned int*)(qp + half * 32 + kb + 2 * j);
                aq[half].d[j + 4] = *(const unsigned int*)(qp + half * 32 + kb + 16 + 2 * j);
            }
    }

    v8f o[4];
#pragma unroll
    for (int i = 0; i < 4; ++i) o[i] = v8f_zero();

    for (int kc = 0; kc < 1024; kc += 32) {
        __syncthreads();   // protect Vs from previous iteration's readers
        // stage V chunk transposed: Vs[d][k]
#pragma unroll
        for (int i = 0; i < 16; ++i) {
            int e = threadIdx.x + i * 128;
            int k = e >> 6, d = e & 63;
            Vs[d][k] = Vbf[(size_t)(b * 1024 + kc + k) * 1024 + h * 64 + d];
        }
        __syncthreads();

        // S tiles for k..k+15 and k+16..k+31, normalized -> Ps (bf16)
#pragma unroll
        for (int kt = 0; kt < 2; ++kt) {
            int kcol = kc + kt * 16 + lo;
            BF16Frag bk[2];
            const unsigned short* kp = Kbf + (size_t)(b * 1024 + kcol) * 1024 + h * 64;
            int kb = hi * 8;
#pragma unroll
            for (int half = 0; half < 2; ++half)
#pragma unroll
                for (int j = 0; j < 8; ++j)
                    bk[half].d[j] = *(const unsigned int*)(kp + half * 32 + kb + 2 * j + (j >> 2) * 8);
            v8f c = v8f_zero();
            c = wmma_bf16(aq[0], bk[0], c);
            c = wmma_bf16(aq[1], bk[1], c);

            float mv  = mbuf[(size_t)bh * 1024 + kcol];
            float inv = 1.0f / sbuf[(size_t)bh * 1024 + kcol];
#pragma unroll
            for (int j = 0; j < 8; ++j) {
                float p = __expf(c[j] * scale - mv) * inv;
                Ps[w][hi * 8 + j][kt * 16 + lo] = f2bf(p);
            }
        }
        __syncthreads();

        // repack P as A-fragment; V fragments from LDS; accumulate O
        BF16Frag ap;
        {
            int kb = hi * 8;
#pragma unroll
            for (int j = 0; j < 4; ++j) {
                ap.d[j]     = *(const unsigned int*)&Ps[w][lo][kb + 2 * j];
                ap.d[j + 4] = *(const unsigned int*)&Ps[w][lo][kb + 16 + 2 * j];
            }
        }
#pragma unroll
        for (int dt = 0; dt < 4; ++dt) {
            BF16Frag bv;
            int d  = dt * 16 + lo;
            int kb = hi * 8;
#pragma unroll
            for (int j = 0; j < 8; ++j)
                bv.d[j] = *(const unsigned int*)&Vs[d][kb + 2 * j + (j >> 2) * 8];
            o[dt] = wmma_bf16(ap, bv, o[dt]);
        }
    }

    // store context (bf16), layout [b, q, h*64+d]
#pragma unroll
    for (int dt = 0; dt < 4; ++dt) {
        int d = h * 64 + dt * 16 + lo;
#pragma unroll
        for (int r = 0; r < 8; ++r) {
            int q = q0 + hi * 8 + r;
            ctx[(size_t)(b * 1024 + q) * 1024 + d] = f2bf(o[dt][r]);
        }
    }
}

// ---------------------------------------------------------------------------
// Host launcher
// ---------------------------------------------------------------------------
extern "C" void kernel_launch(void* const* d_in, const int* in_sizes, int n_in,
                              void* d_out, int out_size, void* d_ws, size_t ws_size,
                              hipStream_t stream) {
    const float* x_q = (const float*)d_in[0];
    const float* x_k = (const float*)d_in[1];
    const float* x_v = (const float*)d_in[2];
    // d_in[3] = mask: reference never applies it
    const float* Wq = (const float*)d_in[4];
    const float* bq = (const float*)d_in[5];
    const float* Wk = (const float*)d_in[6];
    const float* bk = (const float*)d_in[7];
    const float* Wv = (const float*)d_in[8];
    const float* bv = (const float*)d_in[9];
    const float* Wo = (const float*)d_in[10];
    const float* bo = (const float*)d_in[11];

    const int M = 8 * 1024;   // B*L
    const int N = 1024;       // D
    const int K = 1024;       // D

    char* ws = (char*)d_ws;
    const size_t MB = 1024ull * 1024ull;
    unsigned short* Qbf = (unsigned short*)(ws);
    unsigned short* Kbf = (unsigned short*)(ws + 16 * MB);
    unsigned short* Vbf = (unsigned short*)(ws + 32 * MB);
    unsigned short* ctx = (unsigned short*)(ws + 48 * MB);
    float* mbuf = (float*)(ws + 64 * MB);
    float* sbuf = (float*)(ws + 64 * MB + 512 * 1024);

    dim3 blkG(256);
    dim3 gridG(N / 128, M / 128);        // (8, 64)
    dim3 blkA(128);
    dim3 gridA(1024 / 64, 8 * 16);       // (16, 128)

    // Projections (fp32 in, bf16 out)
    gemm_bias_kernel<true,  false><<<gridG, blkG, 0, stream>>>(x_q, Wq, bq, Qbf, M, N, K);
    gemm_bias_kernel<true,  false><<<gridG, blkG, 0, stream>>>(x_k, Wk, bk, Kbf, M, N, K);
    gemm_bias_kernel<true,  false><<<gridG, blkG, 0, stream>>>(x_v, Wv, bv, Vbf, M, N, K);

    // Column-softmax stats, then attention output
    attn_stats_kernel<<<gridA, blkA, 0, stream>>>(Qbf, Kbf, mbuf, sbuf);
    attn_out_kernel<<<gridA, blkA, 0, stream>>>(Qbf, Kbf, Vbf, mbuf, sbuf, ctx);

    // Output projection (bf16 in via async-LDS staging, fp32 out)
    gemm_bias_kernel<false, true><<<gridG, blkG, 0, stream>>>(ctx, Wo, bo, (float*)d_out, M, N, K);
}
